// TrigramAttention_73289321939515
// MI455X (gfx1250) — matrix operand order repaired and checked by reading, not compiled, not run on hardware
//
#include <hip/hip_runtime.h>

typedef __attribute__((ext_vector_type(16))) _Float16 v16h;
typedef __attribute__((ext_vector_type(8)))  _Float16 v8h;
typedef __attribute__((ext_vector_type(2)))  _Float16 v2h;
typedef __attribute__((ext_vector_type(8)))  float    v8f;

#define N_EMBD 1024
#define N_HEAD 16
#define HEAD   64
#define TSEQ   4096
#define BATCH  8
#define MROWS  (BATCH * TSEQ)   /* 32768 */

// ---------------------------------------------------------------------------
// fp32 -> fp16 elementwise convert (grid-stride)
// ---------------------------------------------------------------------------
__global__ void cvt_f32_to_f16(const float* __restrict__ s,
                               _Float16* __restrict__ d, int n) {
  int i = blockIdx.x * blockDim.x + threadIdx.x;
  int stride = gridDim.x * blockDim.x;
  for (; i < n; i += stride) d[i] = (_Float16)s[i];
}

// ---------------------------------------------------------------------------
// WMMA GEMM: out[m,n] = sum_k A[m,k] * W[n,k] + bias[n]   (A @ W^T + b)
//   A: [M, K] f16 row-major;  W: [N, K] f16 row-major
//
// Block = 256 threads = 8 waves arranged 4(M) x 2(N); block tile 128x128.
// Each wave: 32(M) x 64(N) strip = 2 A-frags x 4 B-frags
//   -> 8 v_wmma per K-step of 32, 12 global_load_b128 per K-step.
//
// Fragment layouts used by this code (wave32, same contiguous k order for A and B):
//   A 16x32 f16: lane L(0-15): M=L, a[0..15]=K(kb..kb+15)
//                lane L(16-31): M=L-16, a[0..15]=K(kb+16..31)
//   B 32x16 f16: lanes 0-15: N=lane, b[i]=B[kb+i, N]; lanes 16-31: b[i]=B[kb+16+i, N]
//     with B[k,n] = W[n,k] -> 16 contiguous halves of W row n.
//   C/D 16x16 f32: VGPR r, lane l -> (M = r + 8*(l>>4), N = l&15)
// ---------------------------------------------------------------------------
template <bool OUT_F16>
__global__ __launch_bounds__(256)
void gemm_wmma(const _Float16* __restrict__ A,
               const _Float16* __restrict__ W,
               const float* __restrict__ bias,
               void* __restrict__ outv,
               int K, int ldN) {
  const int lane  = threadIdx.x & 31;
  const int wave  = threadIdx.x >> 5;
  const int waveM = wave >> 1;     // 0..3
  const int waveN = wave & 1;      // 0..1
  const int hl    = lane >> 4;     // 0 or 1 (lane half)
  const int l16   = lane & 15;

  const int mW = blockIdx.y * 128 + waveM * 32;  // wave's M base (2 x 16 rows)
  const int nW = blockIdx.x * 128 + waveN * 64;  // wave's N base (4 x 16 cols)

  const _Float16* ap0 = A + (size_t)(mW + l16) * K + hl * 16;
  const _Float16* ap1 = A + (size_t)(mW + 16 + l16) * K + hl * 16;

  const _Float16* wp[4];
#pragma unroll
  for (int nt = 0; nt < 4; ++nt)
    wp[nt] = W + (size_t)(nW + nt * 16 + l16) * K + hl * 16;

  v8f acc[2][4];
#pragma unroll
  for (int nt = 0; nt < 4; ++nt) {
    float bv = bias[nW + nt * 16 + l16];
#pragma unroll
    for (int r = 0; r < 8; ++r) { acc[0][nt][r] = bv; acc[1][nt][r] = bv; }
  }

  for (int k = 0; k < K; k += 32) {
    v8h a00 = *(const v8h*)(ap0 + k);
    v8h a01 = *(const v8h*)(ap0 + k + 8);
    v16h a0 = __builtin_shufflevector(a00, a01, 0, 1, 2, 3, 4, 5, 6, 7,
                                      8, 9, 10, 11, 12, 13, 14, 15);
    v8h a10 = *(const v8h*)(ap1 + k);
    v8h a11 = *(const v8h*)(ap1 + k + 8);
    v16h a1 = __builtin_shufflevector(a10, a11, 0, 1, 2, 3, 4, 5, 6, 7,
                                      8, 9, 10, 11, 12, 13, 14, 15);
#pragma unroll
    for (int nt = 0; nt < 4; ++nt) {
      v16h b = *(const v16h*)(wp[nt] + k);  // 16 contiguous halves of W row n
      acc[0][nt] = __builtin_amdgcn_wmma_f32_16x16x32_f16(
          false, a0, false, b, (short)0, acc[0][nt], false, false);
      acc[1][nt] = __builtin_amdgcn_wmma_f32_16x16x32_f16(
          false, a1, false, b, (short)0, acc[1][nt], false, false);
    }
  }

#pragma unroll
  for (int mt = 0; mt < 2; ++mt) {
#pragma unroll
    for (int nt = 0; nt < 4; ++nt) {
      const int col = nW + nt * 16 + l16;
#pragma unroll
      for (int r = 0; r < 8; ++r) {
        const int row = mW + mt * 16 + hl * 8 + r;
        if (OUT_F16)
          ((_Float16*)outv)[(size_t)row * ldN + col] = (_Float16)acc[mt][nt][r];
        else
          ((float*)outv)[(size_t)row * ldN + col] = acc[mt][nt][r];
      }
    }
  }
}

// ---------------------------------------------------------------------------
// Banded (trigram) attention. qkv: [MROWS, 3*N_EMBD] f16 (q | k | v blocks).
// One wave32 per (token, head): lanes split d=64 two-per-lane.
// s_j = (q . k_{t-j}) / sqrt(d), softmax over valid j in {0,1,2},
// y = sum_j att_j * v_{t-j}  -> f16 [MROWS, N_EMBD]
// ---------------------------------------------------------------------------
__global__ __launch_bounds__(256)
void band_attn(const _Float16* __restrict__ qkv, _Float16* __restrict__ y) {
  const int lane = threadIdx.x & 31;
  const int wave = threadIdx.x >> 5;
  const int wg = blockIdx.x * 8 + wave;      // [0, MROWS*N_HEAD)
  const int h = wg & (N_HEAD - 1);
  const int m = wg >> 4;                     // global token row
  const int t = m & (TSEQ - 1);              // position within batch element

  const size_t rs = 3 * N_EMBD;              // qkv row stride
  const _Float16* qp = qkv + (size_t)m * rs + h * HEAD + 2 * lane;
  const _Float16* kp = qp + N_EMBD;
  const _Float16* vp = qp + 2 * N_EMBD;

  const v2h q = *(const v2h*)qp;
  const float scale = 0.125f;                // 1/sqrt(64)

  float s[3];
#pragma unroll
  for (int j = 0; j < 3; ++j) {
    float p = 0.0f;
    if (t >= j) {
      v2h kk = *(const v2h*)(kp - (size_t)j * rs);
      p = (float)q[0] * (float)kk[0] + (float)q[1] * (float)kk[1];
    }
#pragma unroll
    for (int off = 16; off >= 1; off >>= 1) p += __shfl_xor(p, off, 32);
    s[j] = p * scale;
  }

  float mx = s[0];
  if (t >= 1 && s[1] > mx) mx = s[1];
  if (t >= 2 && s[2] > mx) mx = s[2];
  const float e0 = __expf(s[0] - mx);
  const float e1 = (t >= 1) ? __expf(s[1] - mx) : 0.0f;
  const float e2 = (t >= 2) ? __expf(s[2] - mx) : 0.0f;
  const float inv = 1.0f / (e0 + e1 + e2);

  v2h v0 = *(const v2h*)vp;
  float y0 = e0 * (float)v0[0];
  float y1 = e0 * (float)v0[1];
  if (t >= 1) {
    v2h vv = *(const v2h*)(vp - rs);
    y0 += e1 * (float)vv[0]; y1 += e1 * (float)vv[1];
  }
  if (t >= 2) {
    v2h vv = *(const v2h*)(vp - 2 * rs);
    y0 += e2 * (float)vv[0]; y1 += e2 * (float)vv[1];
  }
  y0 *= inv; y1 *= inv;

  _Float16* yp = y + (size_t)m * N_EMBD + h * HEAD + 2 * lane;
  yp[0] = (_Float16)y0;
  yp[1] = (_Float16)y1;
}

// ---------------------------------------------------------------------------
extern "C" void kernel_launch(void* const* d_in, const int* in_sizes, int n_in,
                              void* d_out, int out_size, void* d_ws, size_t ws_size,
                              hipStream_t stream) {
  const float* x  = (const float*)d_in[0];
  const float* Wq = (const float*)d_in[1];
  const float* bq = (const float*)d_in[2];
  const float* Wk = (const float*)d_in[3];
  const float* bk = (const float*)d_in[4];
  const float* Wv = (const float*)d_in[5];
  const float* bv = (const float*)d_in[6];
  const float* Wo = (const float*)d_in[7];
  const float* bo = (const float*)d_in[8];

  const int C = N_EMBD;
  char* ws = (char*)d_ws;
  size_t off = 0;
  _Float16* Xh    = (_Float16*)(ws + off); off += (size_t)MROWS * C * 2;      // 64 MB
  _Float16* Wqkvh = (_Float16*)(ws + off); off += (size_t)3 * C * C * 2;      //  6 MB
  _Float16* Woh   = (_Float16*)(ws + off); off += (size_t)C * C * 2;          //  2 MB
  float*    bqkv  = (float*)   (ws + off); off += (size_t)3 * C * 4;          // 12 KB
  _Float16* QKVh  = (_Float16*)(ws + off); off += (size_t)MROWS * 3 * C * 2;  // 192 MB
  _Float16* Yh    = (_Float16*)(ws + off); off += (size_t)MROWS * C * 2;      // 64 MB

  // 1) convert inputs / weights to fp16 in workspace
  cvt_f32_to_f16<<<4096, 256, 0, stream>>>(x, Xh, MROWS * C);
  cvt_f32_to_f16<<<1024, 256, 0, stream>>>(Wq, Wqkvh, C * C);
  cvt_f32_to_f16<<<1024, 256, 0, stream>>>(Wk, Wqkvh + (size_t)C * C, C * C);
  cvt_f32_to_f16<<<1024, 256, 0, stream>>>(Wv, Wqkvh + (size_t)2 * C * C, C * C);
  cvt_f32_to_f16<<<1024, 256, 0, stream>>>(Wo, Woh, C * C);
  hipMemcpyAsync(bqkv,         bq, C * sizeof(float), hipMemcpyDeviceToDevice, stream);
  hipMemcpyAsync(bqkv + C,     bk, C * sizeof(float), hipMemcpyDeviceToDevice, stream);
  hipMemcpyAsync(bqkv + 2 * C, bv, C * sizeof(float), hipMemcpyDeviceToDevice, stream);

  // 2) fused QKV projection: [32768,1024] @ [3072,1024]^T + bqkv -> f16 [32768,3072]
  gemm_wmma<true><<<dim3(3 * C / 128, MROWS / 128), 256, 0, stream>>>(
      Xh, Wqkvh, bqkv, (void*)QKVh, C, 3 * C);

  // 3) banded trigram attention -> f16 [32768,1024]
  band_attn<<<(MROWS * N_HEAD) / 8, 256, 0, stream>>>(QKVh, Yh);

  // 4) output projection: [32768,1024] @ [1024,1024]^T + bo -> f32 d_out
  gemm_wmma<false><<<dim3(C / 128, MROWS / 128), 256, 0, stream>>>(
      Yh, Woh, bo, d_out, C, C);
}
